// PointNet2Seg_50714973831838
// MI455X (gfx1250) — matrix operand order repaired and checked by reading, not compiled, not run on hardware
//
#include <hip/hip_runtime.h>
#include <hip/hip_bf16.h>

typedef float v8f __attribute__((ext_vector_type(8)));
typedef __bf16 v16bf __attribute__((ext_vector_type(16)));
typedef unsigned short us16 __attribute__((ext_vector_type(16)));

__device__ __forceinline__ unsigned short f2bf(float f) {
  unsigned int u = __float_as_uint(f);
  unsigned int r = u + 0x7fffu + ((u >> 16) & 1u);   // round-to-nearest-even
  return (unsigned short)(r >> 16);
}

// ---------------------------------------------------------------------------
// Fused GEMM: Y[M,Cout] = relu_bn( X[M,K] * W[Cout,K]^T + b )   (bf16 WMMA)
// fuse==1: y = relu((acc + b)*g/sqrt(1+eps) + beta);  fuse==0: y = acc + b
// ---------------------------------------------------------------------------
#define TM 128
#define TN 64
#define TK 32

__global__ void __launch_bounds__(256) k_gemm(
    const float* __restrict__ X, int M, int Kd,
    const float* __restrict__ W, const float* __restrict__ bias,
    const float* __restrict__ gamma, const float* __restrict__ beta,
    float* __restrict__ Y, int Cout, int fuse)
{
  __shared__ __align__(16) unsigned short As[TM][TK + 4];
  __shared__ __align__(16) unsigned short Bs[TN][TK + 4];

  const int tid  = threadIdx.x;
  const int m0   = blockIdx.x * TM;
  const int n0   = blockIdx.y * TN;
  const int wave = tid >> 5;
  const int lane = tid & 31;
  const int wm   = (wave & 3) << 5;   // 4 waves along M (32 rows each)
  const int wn   = (wave >> 2) << 5;  // 2 waves along N (32 cols each)

  const bool fullM = (m0 + TM <= M);
  const bool fullN = (n0 + TN <= Cout);

  v8f acc[2][2] = {};

  for (int k0 = 0; k0 < Kd; k0 += TK) {
    const bool fullK = (k0 + TK <= Kd);

    if (fullM && fullK) {
      // fast path: unpredicated vector staging; 4 consecutive K elems/thread
      #pragma unroll
      for (int it = 0; it < 4; ++it) {
        int g  = tid + it * 256;          // 1024 groups of 4 over 128x32
        int r  = g >> 3;
        int c4 = (g & 7) << 2;
        const float* src = X + (size_t)(m0 + r) * Kd + (k0 + c4);
        float a0 = src[0], a1 = src[1], a2 = src[2], a3 = src[3];
        __builtin_prefetch(src + TK, 0, 0);   // global_prefetch_b8 of next K-tile
        uint2 pk;
        pk.x = (unsigned)f2bf(a0) | ((unsigned)f2bf(a1) << 16);
        pk.y = (unsigned)f2bf(a2) | ((unsigned)f2bf(a3) << 16);
        *(uint2*)&As[r][c4] = pk;            // one ds_store_b64
      }
    } else {
      for (int i = tid; i < TM * TK; i += 256) {
        int r = i >> 5, c = i & 31;
        int gm = m0 + r, gk = k0 + c;
        float v = (gm < M && gk < Kd) ? X[(size_t)gm * Kd + gk] : 0.0f;
        As[r][c] = f2bf(v);
      }
    }

    if (fullN && fullK) {
      #pragma unroll
      for (int it = 0; it < 2; ++it) {
        int g  = tid + it * 256;          // 512 groups of 4 over 64x32
        int r  = g >> 3;
        int c4 = (g & 7) << 2;
        const float* src = W + (size_t)(n0 + r) * Kd + (k0 + c4);
        float a0 = src[0], a1 = src[1], a2 = src[2], a3 = src[3];
        uint2 pk;
        pk.x = (unsigned)f2bf(a0) | ((unsigned)f2bf(a1) << 16);
        pk.y = (unsigned)f2bf(a2) | ((unsigned)f2bf(a3) << 16);
        *(uint2*)&Bs[r][c4] = pk;
      }
    } else {
      for (int i = tid; i < TN * TK; i += 256) {
        int r = i >> 5, c = i & 31;
        int gn = n0 + r, gk = k0 + c;
        float v = (gn < Cout && gk < Kd) ? W[(size_t)gn * Kd + gk] : 0.0f;
        Bs[r][c] = f2bf(v);
      }
    }
    __syncthreads();

    // A fragment: 16x32 bf16, lane L holds row M=L%16; K per ISA layout
    us16 av[2], bv[2];
    const int arow = lane & 15;
    const int akb  = (lane >> 4) * 8;
    #pragma unroll
    for (int f = 0; f < 2; ++f) {
      us16 t;
      const int row = wm + f * 16 + arow;
      #pragma unroll
      for (int g = 0; g < 2; ++g)
        #pragma unroll
        for (int j = 0; j < 8; ++j)
          t[g * 8 + j] = As[row][akb + g * 16 + j];
      av[f] = t;
    }
    // B fragment: 32x16 bf16, lane L holds col N=L%16, K = (L/16)*8 + (h/8)*16 + h%8
    const int bcol = lane & 15;
    const int bkb  = (lane >> 4) * 8;
    #pragma unroll
    for (int f = 0; f < 2; ++f) {
      us16 t;
      const int col = wn + f * 16 + bcol;
      #pragma unroll
      for (int h = 0; h < 16; ++h)
        t[h] = Bs[col][bkb + (h >> 3) * 16 + (h & 7)];
      bv[f] = t;
    }

    #pragma unroll
    for (int i = 0; i < 2; ++i)
      #pragma unroll
      for (int j = 0; j < 2; ++j)
        acc[i][j] = __builtin_amdgcn_wmma_f32_16x16x32_bf16(
            false, __builtin_bit_cast(v16bf, av[i]),
            false, __builtin_bit_cast(v16bf, bv[j]),
            (short)0, acc[i][j], false, false);
    __syncthreads();
  }

  // Epilogue: D layout — vgpr r, lane L -> m = r + (L/16)*8, n = L%16
  const float rs = rsqrtf(1.0f + 1e-5f);
  #pragma unroll
  for (int i = 0; i < 2; ++i) {
    #pragma unroll
    for (int j = 0; j < 2; ++j) {
      int gn = n0 + wn + j * 16 + (lane & 15);
      if (gn >= Cout) continue;
      float bb = bias[gn];
      float s = 0.f, sh = 0.f;
      if (fuse) { s = gamma[gn] * rs; sh = bb * s + beta[gn]; }
      #pragma unroll
      for (int r = 0; r < 8; ++r) {
        int gm = m0 + wm + i * 16 + r + ((lane >> 4) * 8);
        if (gm >= M) continue;
        float v = acc[i][j][r];
        v = fuse ? fmaxf(v * s + sh, 0.0f) : (v + bb);
        Y[(size_t)gm * Cout + gn] = v;
      }
    }
  }
}

// ---------------------------------------------------------------------------
// [B,C,N] -> [B,N,C]
__global__ void k_transpose_in(const float* __restrict__ x, float* __restrict__ xyz,
                               int Bn, int C, int Nn) {
  int t = blockIdx.x * 256 + threadIdx.x;
  if (t >= Bn * Nn) return;
  int b = t / Nn, n = t - b * Nn;
  for (int c = 0; c < C; ++c)
    xyz[(size_t)t * C + c] = x[((size_t)b * C + c) * Nn + n];
}

// Farthest point sampling; one block per batch. dist = global scratch [B,Nn].
__global__ void __launch_bounds__(1024) k_fps(const float* __restrict__ xyz,
                                              float* __restrict__ dist,
                                              int Nn, int npoint, int* __restrict__ out) {
  const int b = blockIdx.x, tid = threadIdx.x;
  const float* P = xyz + (size_t)b * Nn * 3;
  float* D = dist + (size_t)b * Nn;
  __shared__ float rv[1024];
  __shared__ int   ri[1024];
  __shared__ int   s_far;
  for (int i = tid; i < Nn; i += 1024) D[i] = 1e10f;
  if (tid == 0) s_far = 0;
  __syncthreads();
  for (int it = 0; it < npoint; ++it) {
    int far = s_far;
    if (tid == 0) out[b * npoint + it] = far;
    float cx = P[far * 3 + 0], cy = P[far * 3 + 1], cz = P[far * 3 + 2];
    float bestv = -1.0f; int besti = 0x7fffffff;
    for (int i = tid; i < Nn; i += 1024) {
      float dx = P[i * 3 + 0] - cx, dy = P[i * 3 + 1] - cy, dz = P[i * 3 + 2] - cz;
      float d = fminf(D[i], dx * dx + dy * dy + dz * dz);
      D[i] = d;
      if (d > bestv) { bestv = d; besti = i; }
    }
    rv[tid] = bestv; ri[tid] = besti;
    __syncthreads();
    for (int off = 512; off > 0; off >>= 1) {
      if (tid < off) {
        float ov = rv[tid + off]; int oi = ri[tid + off];
        if (ov > rv[tid] || (ov == rv[tid] && oi < ri[tid])) { rv[tid] = ov; ri[tid] = oi; }
      }
      __syncthreads();
    }
    if (tid == 0) s_far = ri[0];
    __syncthreads();
  }
}

// dst[b,s,:] = src[b, idx[b,s], :]
__global__ void k_gather(const float* __restrict__ src, const int* __restrict__ idx,
                         float* __restrict__ dst, int Bn, int Nn, int S, int C) {
  int t = blockIdx.x * 256 + threadIdx.x;
  if (t >= Bn * S) return;
  int b = t / S;
  int id = idx[t];
  const float* s = src + ((size_t)b * Nn + id) * C;
  float* d = dst + (size_t)t * C;
  for (int c = 0; c < C; ++c) d[c] = s[c];
}

// 32-NN (sorted ascending, earlier index wins ties)
__global__ void k_knn(const float* __restrict__ q, const float* __restrict__ ref,
                      int* __restrict__ out, int Bn, int S, int Nn) {
  int t = blockIdx.x * 256 + threadIdx.x;
  if (t >= Bn * S) return;
  int b = t / S;
  const float* Q = q + (size_t)t * 3;
  const float* R = ref + (size_t)b * Nn * 3;
  float qx = Q[0], qy = Q[1], qz = Q[2];
  float bd[32]; int bi[32];
  for (int i = 0; i < 32; ++i) { bd[i] = 3.4e38f; bi[i] = 0; }
  for (int n = 0; n < Nn; ++n) {
    float dx = R[n * 3 + 0] - qx, dy = R[n * 3 + 1] - qy, dz = R[n * 3 + 2] - qz;
    float d = dx * dx + dy * dy + dz * dz;
    if (d < bd[31]) {
      int j = 31;
      while (j > 0 && d < bd[j - 1]) { bd[j] = bd[j - 1]; bi[j] = bi[j - 1]; --j; }
      bd[j] = d; bi[j] = n;
    }
  }
  int* O = out + (size_t)t * 32;
  for (int i = 0; i < 32; ++i) O[i] = bi[i];
}

// grouped feature rows: [centered xyz (3) | gathered point feats (C)]
__global__ void k_group(const float* __restrict__ xyz, const float* __restrict__ pts,
                        const float* __restrict__ new_xyz, const int* __restrict__ idx,
                        float* __restrict__ out, int Bn, int Nn, int S, int k, int C) {
  int t = blockIdx.x * 256 + threadIdx.x;
  if (t >= Bn * S * k) return;
  int bs = t / k;
  int b = bs / S;
  int id = idx[t];
  const float* p = xyz + ((size_t)b * Nn + id) * 3;
  const float* c = new_xyz + (size_t)bs * 3;
  float* o = out + (size_t)t * (3 + C);
  o[0] = p[0] - c[0]; o[1] = p[1] - c[1]; o[2] = p[2] - c[2];
  const float* f = pts + ((size_t)b * Nn + id) * C;
  for (int cc = 0; cc < C; ++cc) o[3 + cc] = f[cc];
}

__global__ void k_maxpool(const float* __restrict__ in, float* __restrict__ out,
                          int rows, int k, int C) {
  int t = blockIdx.x * 256 + threadIdx.x;
  if (t >= rows * C) return;
  int r = t / C, c = t - r * C;
  float m = -3.4e38f;
  const float* p = in + ((size_t)r * k) * C + c;
  for (int j = 0; j < k; ++j) m = fmaxf(m, p[(size_t)j * C]);
  out[t] = m;
}

// out[r] = [A[r] (Ca) | Bp[r or r/nperb] (Cb)]
__global__ void k_concat2(const float* __restrict__ A, int Ca,
                          const float* __restrict__ Bp, int Cb,
                          float* __restrict__ out, int rows, int bcast_nperb) {
  int C = Ca + Cb;
  size_t total = (size_t)rows * C;
  size_t t = (size_t)blockIdx.x * 256 + threadIdx.x;
  if (t >= total) return;
  int r = (int)(t / C), c = (int)(t - (size_t)r * C);
  float v;
  if (c < Ca) v = A[(size_t)r * Ca + c];
  else {
    int rb = (bcast_nperb > 0) ? (r / bcast_nperb) : r;
    v = Bp[(size_t)rb * Cb + (c - Ca)];
  }
  out[t] = v;
}

// 3-NN inverse-distance interpolation
__global__ void k_interp3(const float* __restrict__ tq, const float* __restrict__ sx,
                          const float* __restrict__ sf, float* __restrict__ out,
                          int Bn, int Nt, int S, int C) {
  int t = blockIdx.x * 256 + threadIdx.x;
  if (t >= Bn * Nt) return;
  int b = t / Nt;
  const float* Q = tq + (size_t)t * 3;
  const float* X = sx + (size_t)b * S * 3;
  float qx = Q[0], qy = Q[1], qz = Q[2];
  float d0 = 3.4e38f, d1 = 3.4e38f, d2 = 3.4e38f;
  int i0 = 0, i1 = 0, i2 = 0;
  for (int s = 0; s < S; ++s) {
    float dx = X[s * 3] - qx, dy = X[s * 3 + 1] - qy, dz = X[s * 3 + 2] - qz;
    float d = dx * dx + dy * dy + dz * dz;
    if (d < d0)      { d2 = d1; i2 = i1; d1 = d0; i1 = i0; d0 = d; i0 = s; }
    else if (d < d1) { d2 = d1; i2 = i1; d1 = d;  i1 = s; }
    else if (d < d2) { d2 = d;  i2 = s; }
  }
  float w0 = 1.f / (d0 + 1e-8f), w1 = 1.f / (d1 + 1e-8f), w2 = 1.f / (d2 + 1e-8f);
  float wsum = w0 + w1 + w2;
  w0 /= wsum; w1 /= wsum; w2 /= wsum;
  const float* F = sf + (size_t)b * S * C;
  float* O = out + (size_t)t * C;
  for (int c = 0; c < C; ++c)
    O[c] = w0 * F[(size_t)i0 * C + c] + w1 * F[(size_t)i1 * C + c] + w2 * F[(size_t)i2 * C + c];
}

// Y[(b*N+n)*C + c] -> out[(b*C+c)*N + n]
__global__ void k_transpose_out(const float* __restrict__ Y, float* __restrict__ out,
                                int Bn, int Nn, int C) {
  int t = blockIdx.x * 256 + threadIdx.x;
  if (t >= Bn * Nn * C) return;
  int c = t % C; int bn = t / C; int n = bn % Nn; int b = bn / Nn;
  out[((size_t)b * C + c) * Nn + n] = Y[t];
}

// ---------------------------------------------------------------------------
struct LayerP { const float *W, *b, *g, *beta; int cin, cout; };

extern "C" void kernel_launch(void* const* d_in, const int* in_sizes, int n_in,
                              void* d_out, int out_size, void* d_ws, size_t ws_size,
                              hipStream_t stream) {
  (void)in_sizes; (void)n_in; (void)out_size; (void)ws_size;
  const int B = 4, N0 = 16384, S1 = 1024, S2 = 256, KNN = 32, NC = 13;

  // --- inputs (setup_inputs dict insertion order) ---
  int pi = 0;
  const float* x = (const float*)d_in[pi++];
  auto takeBN = [&](int cin, int cout) -> LayerP {
    LayerP L;
    L.W = (const float*)d_in[pi++]; L.b = (const float*)d_in[pi++];
    L.g = (const float*)d_in[pi++]; L.beta = (const float*)d_in[pi++];
    L.cin = cin; L.cout = cout; return L;
  };
  LayerP sa1_0 = takeBN(6, 64),    sa1_1 = takeBN(64, 64),   sa1_2 = takeBN(64, 128);
  LayerP sa2_0 = takeBN(131, 128), sa2_1 = takeBN(128, 128), sa2_2 = takeBN(128, 256);
  LayerP sa3_0 = takeBN(259, 256), sa3_1 = takeBN(256, 512), sa3_2 = takeBN(512, 1024);
  LayerP fp3_0 = takeBN(1280, 256), fp3_1 = takeBN(256, 256);
  LayerP fp2_0 = takeBN(384, 256),  fp2_1 = takeBN(256, 128);
  LayerP fp1_0 = takeBN(131, 128),  fp1_1 = takeBN(128, 128), fp1_2 = takeBN(128, 128);
  LayerP head1 = takeBN(128, 128);
  const float* h2W = (const float*)d_in[pi++];
  const float* h2b = (const float*)d_in[pi++];

  // --- workspace bump allocator ---
  char* ws = (char*)d_ws;
  size_t off = 0;
  auto alloc = [&](size_t bytes) -> void* {
    void* r = ws + off;
    off += (bytes + 255) & ~(size_t)255;
    return r;
  };
  float* xyz0    = (float*)alloc((size_t)B * N0 * 3 * 4);
  float* dist_ws = (float*)alloc((size_t)B * N0 * 4);
  int*   fidx1   = (int*)alloc((size_t)B * S1 * 4);
  float* xyz1    = (float*)alloc((size_t)B * S1 * 3 * 4);
  int*   fidx2   = (int*)alloc((size_t)B * S2 * 4);
  float* xyz2    = (float*)alloc((size_t)B * S2 * 3 * 4);
  int*   idx1    = (int*)alloc((size_t)B * S1 * KNN * 4);
  int*   idx2    = (int*)alloc((size_t)B * S2 * KNN * 4);
  float* p1      = (float*)alloc((size_t)B * S1 * 128 * 4);
  float* p2      = (float*)alloc((size_t)B * S2 * 256 * 4);
  float* p3      = (float*)alloc((size_t)B * 1024 * 4);
  float* p2n     = (float*)alloc((size_t)B * S2 * 256 * 4);
  float* p1n     = (float*)alloc((size_t)B * S1 * 128 * 4);
  float* interp2 = (float*)alloc((size_t)B * S1 * 256 * 4);
  float* interp1 = (float*)alloc((size_t)B * N0 * 128 * 4);
  float* pp0     = (float*)alloc((size_t)B * S1 * KNN * 128 * 4);   // 131072 x 128
  float* pp1     = (float*)alloc((size_t)B * S1 * KNN * 128 * 4);

  auto blocks = [](size_t total) { return dim3((unsigned)((total + 255) / 256)); };
  auto gemm = [&](const float* X, int M, const LayerP& L, float* Y, int fuse) {
    dim3 g((M + TM - 1) / TM, (L.cout + TN - 1) / TN);
    k_gemm<<<g, 256, 0, stream>>>(X, M, L.cin, L.W, L.b, L.g, L.beta, Y, L.cout, fuse);
  };

  // --- stage 0: transpose input (pts0 == xyz0 since C==3) ---
  k_transpose_in<<<blocks((size_t)B * N0), 256, 0, stream>>>(x, xyz0, B, 3, N0);

  // --- SA1 ---
  k_fps<<<B, 1024, 0, stream>>>(xyz0, dist_ws, N0, S1, fidx1);
  k_gather<<<blocks((size_t)B * S1), 256, 0, stream>>>(xyz0, fidx1, xyz1, B, N0, S1, 3);
  k_knn<<<blocks((size_t)B * S1), 256, 0, stream>>>(xyz1, xyz0, idx1, B, S1, N0);
  k_group<<<blocks((size_t)B * S1 * KNN), 256, 0, stream>>>(xyz0, xyz0, xyz1, idx1, pp0,
                                                            B, N0, S1, KNN, 3);
  int M1 = B * S1 * KNN;                      // 131072 rows, Cin=6
  gemm(pp0, M1, sa1_0, pp1, 1);
  gemm(pp1, M1, sa1_1, pp0, 1);
  gemm(pp0, M1, sa1_2, pp1, 1);
  k_maxpool<<<blocks((size_t)B * S1 * 128), 256, 0, stream>>>(pp1, p1, B * S1, KNN, 128);

  // --- SA2 ---
  k_fps<<<B, 1024, 0, stream>>>(xyz1, dist_ws, S1, S2, fidx2);
  k_gather<<<blocks((size_t)B * S2), 256, 0, stream>>>(xyz1, fidx2, xyz2, B, S1, S2, 3);
  k_knn<<<blocks((size_t)B * S2), 256, 0, stream>>>(xyz2, xyz1, idx2, B, S2, S1);
  k_group<<<blocks((size_t)B * S2 * KNN), 256, 0, stream>>>(xyz1, p1, xyz2, idx2, pp0,
                                                            B, S1, S2, KNN, 128);
  int M2 = B * S2 * KNN;                      // 32768 rows, Cin=131
  gemm(pp0, M2, sa2_0, pp1, 1);
  gemm(pp1, M2, sa2_1, pp0, 1);
  gemm(pp0, M2, sa2_2, pp1, 1);
  k_maxpool<<<blocks((size_t)B * S2 * 256), 256, 0, stream>>>(pp1, p2, B * S2, KNN, 256);

  // --- SA3 (group all) ---
  int M3 = B * S2;                            // 1024 rows, Cin=259
  k_concat2<<<blocks((size_t)M3 * 259), 256, 0, stream>>>(xyz2, 3, p2, 256, pp0, M3, 0);
  gemm(pp0, M3, sa3_0, pp1, 1);
  gemm(pp1, M3, sa3_1, pp0, 1);
  gemm(pp0, M3, sa3_2, pp1, 1);
  k_maxpool<<<blocks((size_t)B * 1024), 256, 0, stream>>>(pp1, p3, B, S2, 1024);

  // --- FP3: broadcast p3 onto 256 centers, concat with p2 ---
  k_concat2<<<blocks((size_t)M3 * 1280), 256, 0, stream>>>(p2, 256, p3, 1024, pp0, M3, S2);
  gemm(pp0, M3, fp3_0, pp1, 1);
  gemm(pp1, M3, fp3_1, p2n, 1);

  // --- FP2: interp p2n (256 src) -> 1024 targets, concat with p1 ---
  k_interp3<<<blocks((size_t)B * S1), 256, 0, stream>>>(xyz1, xyz2, p2n, interp2, B, S1, S2, 256);
  int M4 = B * S1;                            // 4096 rows, Cin=384
  k_concat2<<<blocks((size_t)M4 * 384), 256, 0, stream>>>(p1, 128, interp2, 256, pp0, M4, 0);
  gemm(pp0, M4, fp2_0, pp1, 1);
  gemm(pp1, M4, fp2_1, p1n, 1);

  // --- FP1: interp p1n (1024 src) -> 16384 targets, concat with pts0(=xyz0) ---
  k_interp3<<<blocks((size_t)B * N0), 256, 0, stream>>>(xyz0, xyz1, p1n, interp1, B, N0, S1, 128);
  int M5 = B * N0;                            // 65536 rows, Cin=131
  k_concat2<<<blocks((size_t)M5 * 131), 256, 0, stream>>>(xyz0, 3, interp1, 128, pp0, M5, 0);
  gemm(pp0, M5, fp1_0, pp1, 1);
  gemm(pp1, M5, fp1_1, pp0, 1);
  gemm(pp0, M5, fp1_2, pp1, 1);

  // --- heads ---
  gemm(pp1, M5, head1, pp0, 1);
  {
    dim3 g((M5 + TM - 1) / TM, (NC + TN - 1) / TN);
    k_gemm<<<g, 256, 0, stream>>>(pp0, M5, 128, h2W, h2b, nullptr, nullptr, pp1, NC, 0);
  }
  k_transpose_out<<<blocks((size_t)M5 * NC), 256, 0, stream>>>(pp1, (float*)d_out, B, N0, NC);
}
